// ACTBasic_22385369547136
// MI455X (gfx1250) — compile-verified, matrix operand order checked by reading, not run on hardware
//
#include <hip/hip_runtime.h>
#include <hip/hip_bf16.h>

typedef __attribute__((ext_vector_type(16))) __bf16 v16bf;
typedef __attribute__((ext_vector_type(8)))  float  v8f;
typedef __attribute__((ext_vector_type(4)))  int    v4i_;

#if defined(__has_builtin)
#if __has_builtin(__builtin_amdgcn_global_load_async_to_lds_b128)
#define HAVE_ASYNC_LDS 1
#endif
#endif

union FragU { int4 q[2]; v16bf v; };

__device__ inline v16bf lds_frag(const __bf16* rowp, int lhalf) {
    FragU f;
    f.q[0] = *reinterpret_cast<const int4*>(rowp + lhalf * 8);
    f.q[1] = *reinterpret_cast<const int4*>(rowp + lhalf * 8 + 16);
    return f.v;
}

__device__ inline void async_copy_b128(const void* gsrc, void* ldst) {
#if defined(HAVE_ASYNC_LDS)
    __builtin_amdgcn_global_load_async_to_lds_b128(
        (__attribute__((address_space(1))) v4i_*)(uintptr_t)gsrc,
        (__attribute__((address_space(3))) v4i_*)(uint32_t)(uintptr_t)ldst,
        0, 0);
#endif
}

__device__ inline void wait_async_tiles() {
#if defined(HAVE_ASYNC_LDS)
#if __has_builtin(__builtin_amdgcn_s_wait_asynccnt)
    __builtin_amdgcn_s_wait_asynccnt(0);
#else
    asm volatile("s_wait_asynccnt 0x0" ::: "memory");
#endif
#endif
}

// ---------------------------------------------------------------------------
// Prep kernels
// ---------------------------------------------------------------------------
__global__ void init_big_kernel(float* __restrict__ prev, float* __restrict__ st,
                                const float* __restrict__ state_in, int n) {
    int i = blockIdx.x * blockDim.x + threadIdx.x;
    if (i < n) { prev[i] = 0.0f; st[i] = state_in[i]; }
}

__global__ void init_small_kernel(float* hp, float* rem, float* nu, float* uw,
                                  int* flags, int ntok, int nflags) {
    int i = blockIdx.x * blockDim.x + threadIdx.x;
    if (i < ntok) { hp[i] = 0.0f; rem[i] = 0.0f; nu[i] = 0.0f; uw[i] = 0.0f; }
    if (i < nflags) flags[i] = 0;
}

// W [R][C] fp32 row-major  ->  WT [C][R] bf16 row-major (transpose+convert)
__global__ void convert_transpose_kernel(const float* __restrict__ W,
                                         __bf16* __restrict__ WT, int R, int C) {
    __shared__ __bf16 tile[32][33];
    int bx = blockIdx.x, by = blockIdx.y;
    int tx = threadIdx.x, ty = threadIdx.y;   // (32, 8)
#pragma unroll
    for (int i = 0; i < 4; ++i) {
        int r = by * 32 + ty + i * 8;
        int c = bx * 32 + tx;
        tile[ty + i * 8][tx] = (__bf16)W[(size_t)r * C + c];
    }
    __syncthreads();
#pragma unroll
    for (int i = 0; i < 4; ++i) {
        int c = bx * 32 + ty + i * 8;         // output row = original column
        int r = by * 32 + tx;
        WT[(size_t)c * R + r] = tile[tx][ty + i * 8];
    }
}

// ---------------------------------------------------------------------------
// Per-step scalar/gating kernels
// ---------------------------------------------------------------------------
__global__ void gate_kernel(const float* __restrict__ hp, const float* __restrict__ nu,
                            int* __restrict__ flag, int ntok) {
    int i = blockIdx.x * blockDim.x + threadIdx.x;
    if (i < ntok && hp[i] < 0.9f && nu[i] < 6.0f) atomicOr(flag, 1);
}

// wave-per-token: compute s = state+time+pos, p = sigmoid(s.w_p + b_p),
// halting bookkeeping, and emit s as bf16 for the GEMM.
__global__ __launch_bounds__(256)
void p_halt_kernel(const float* __restrict__ state, const float* __restrict__ timesig,
                   const float* __restrict__ pos,   // already offset by step*D
                   const float* __restrict__ w_p,   const float* __restrict__ b_p,
                   float* __restrict__ hp, float* __restrict__ rem,
                   float* __restrict__ nu, float* __restrict__ uwv,
                   __bf16* __restrict__ sbf, const int* __restrict__ flag) {
    if (*flag == 0) return;
    const int wave = threadIdx.x >> 5;
    const int lane = threadIdx.x & 31;
    const int tok  = blockIdx.x * 8 + wave;          // 8192 tokens
    const int q    = tok & 1023;                     // seq position
    const float* sp = state   + (size_t)tok * 1024;
    const float* tp = timesig + (size_t)q   * 1024;
    __bf16* so = sbf + (size_t)tok * 1024;

    float dot = 0.0f;
#pragma unroll
    for (int it = 0; it < 8; ++it) {
        int j = it * 128 + lane * 4;
        float4 sv = *reinterpret_cast<const float4*>(sp  + j);
        float4 tv = *reinterpret_cast<const float4*>(tp  + j);
        float4 pv = *reinterpret_cast<const float4*>(pos + j);
        float4 wv = *reinterpret_cast<const float4*>(w_p + j);
        float s0 = sv.x + tv.x + pv.x;
        float s1 = sv.y + tv.y + pv.y;
        float s2 = sv.z + tv.z + pv.z;
        float s3 = sv.w + tv.w + pv.w;
        dot += s0 * wv.x + s1 * wv.y + s2 * wv.z + s3 * wv.w;
        union { __bf16 b[4]; uint2 u; } pk;
        pk.b[0] = (__bf16)s0; pk.b[1] = (__bf16)s1;
        pk.b[2] = (__bf16)s2; pk.b[3] = (__bf16)s3;
        *reinterpret_cast<uint2*>(so + j) = pk.u;
    }
#pragma unroll
    for (int o = 16; o > 0; o >>= 1) dot += __shfl_xor(dot, o, 32);

    if (lane == 0) {
        float p  = 1.0f / (1.0f + __expf(-(dot + b_p[0])));
        float h  = hp[tok], rm = rem[tok], n = nu[tok];
        float sr  = (h < 1.0f) ? 1.0f : 0.0f;
        float acc = h + p * sr;
        float nh  = (acc >  0.9f) ? sr : 0.0f;
        float sr2 = (acc <= 0.9f) ? sr : 0.0f;
        float hn  = h + p * sr2;
        float rn  = rm + nh * (1.0f - hn);
        hn += nh * rn;
        float nn = n + sr2 + nh;
        float u  = p * sr2 + nh * rn;
        hp[tok] = hn; rem[tok] = rn; nu[tok] = nn; uwv[tok] = u;
    }
}

__global__ void finalize_kernel(const float* __restrict__ rem, const float* __restrict__ nu,
                                float* __restrict__ out_tail, int ntok) {
    int i = blockIdx.x * blockDim.x + threadIdx.x;
    if (i < ntok) { out_tail[i] = rem[i]; out_tail[ntok + i] = nu[i]; }
}

// ---------------------------------------------------------------------------
// bf16 WMMA GEMM: C[M,N] = A[M,K] @ B[K,N], with BT stored as [N][K] bf16.
// 128x128 block tile, BK=32, 8 waves each computing 64x32 (4x2 wmma tiles).
// Tiles staged with GLOBAL_LOAD_ASYNC_TO_LDS_B128 (ASYNCcnt), double-buffered.
// EPI==1: H = bf16(relu(acc + bias))          (GEMM1, N = d_ff)
// EPI==2: s_t = acc + bias; commit prev/state (GEMM2, N = d)
// ---------------------------------------------------------------------------
template <int EPI>
__global__ __launch_bounds__(256)
void gemm_bf16_wmma(const __bf16* __restrict__ A, const __bf16* __restrict__ BT,
                    int M, int N, int K, const float* __restrict__ bias,
                    __bf16* __restrict__ Hout,
                    float* __restrict__ prev, float* __restrict__ state,
                    const float* __restrict__ uw, const int* __restrict__ flag) {
    if (*flag == 0) return;   // ACT early exit: nothing to commit this hop

    __shared__ __align__(16) __bf16 sA[2][128][40];
    __shared__ __align__(16) __bf16 sB[2][128][40];

    const int tid   = threadIdx.x;
    const int lane  = tid & 31;
    const int wave  = tid >> 5;
    const int wm    = wave >> 2;        // 0..1  (M direction, 64 rows each)
    const int wn    = wave & 3;         // 0..3  (N direction, 32 cols each)
    const int lrow  = lane & 15;
    const int lhalf = lane >> 4;

    const int m0 = blockIdx.y * 128;
    const int n0 = blockIdx.x * 128;

#if defined(HAVE_ASYNC_LDS)
    auto load_tile = [&](int b, int k0) {
#pragma unroll
        for (int i = 0; i < 2; ++i) {
            int c  = tid + i * 256;       // 512 chunks of 8 bf16 (16B)
            int r  = c >> 2;
            int kc = (c & 3) * 8;
            async_copy_b128(&A[(size_t)(m0 + r) * K + k0 + kc],  &sA[b][r][kc]);
            async_copy_b128(&BT[(size_t)(n0 + r) * K + k0 + kc], &sB[b][r][kc]);
        }
    };
#else
    auto load_tile = [&](int b, int k0) {
        int4 ta[2], tb[2];
#pragma unroll
        for (int i = 0; i < 2; ++i) {      // batch all global loads first
            int c  = tid + i * 256;
            int r  = c >> 2;
            int kc = (c & 3) * 8;
            ta[i] = *reinterpret_cast<const int4*>(&A[(size_t)(m0 + r) * K + k0 + kc]);
            tb[i] = *reinterpret_cast<const int4*>(&BT[(size_t)(n0 + r) * K + k0 + kc]);
        }
#pragma unroll
        for (int i = 0; i < 2; ++i) {      // then all LDS stores
            int c  = tid + i * 256;
            int r  = c >> 2;
            int kc = (c & 3) * 8;
            *reinterpret_cast<int4*>(&sA[b][r][kc]) = ta[i];
            *reinterpret_cast<int4*>(&sB[b][r][kc]) = tb[i];
        }
    };
#endif

    v8f acc[4][2];
    const v8f vzero = {0.f, 0.f, 0.f, 0.f, 0.f, 0.f, 0.f, 0.f};
#pragma unroll
    for (int i = 0; i < 4; ++i)
#pragma unroll
        for (int j = 0; j < 2; ++j) acc[i][j] = vzero;

    const int nk = K >> 5;
    load_tile(0, 0);
    wait_async_tiles();
    __syncthreads();

    int buf = 0;
    for (int kb = 0; kb < nk; ++kb) {
        if (kb + 1 < nk) load_tile(buf ^ 1, (kb + 1) << 5);

        v16bf af[4], bf[2];
#pragma unroll
        for (int i = 0; i < 4; ++i)
            af[i] = lds_frag(&sA[buf][wm * 64 + i * 16 + lrow][0], lhalf);
#pragma unroll
        for (int j = 0; j < 2; ++j)
            bf[j] = lds_frag(&sB[buf][wn * 32 + j * 16 + lrow][0], lhalf);

#pragma unroll
        for (int i = 0; i < 4; ++i)
#pragma unroll
            for (int j = 0; j < 2; ++j)
                acc[i][j] = __builtin_amdgcn_wmma_f32_16x16x32_bf16(
                    false, af[i], false, bf[j], (short)0, acc[i][j], false, false);

        wait_async_tiles();
        __syncthreads();
        buf ^= 1;
    }

    // Epilogue. C 16x16 f32 layout: VGPR r holds row (r + lhalf*8), col = lane&15.
#pragma unroll
    for (int i = 0; i < 4; ++i) {
#pragma unroll
        for (int j = 0; j < 2; ++j) {
            const int col = n0 + wn * 32 + j * 16 + lrow;
            const float bv = bias[col];
#pragma unroll
            for (int r = 0; r < 8; ++r) {
                const int row = m0 + wm * 64 + i * 16 + r + lhalf * 8;
                float v = acc[i][j][r] + bv;
                if constexpr (EPI == 1) {
                    v = fmaxf(v, 0.0f);
                    Hout[(size_t)row * N + col] = (__bf16)v;
                } else {
                    const float u = uw[row];
                    const size_t idx = (size_t)row * N + col;
                    const float pv = prev[idx];
                    prev[idx]  = v * u + pv * (1.0f - u);
                    state[idx] = v;
                }
            }
        }
    }
}

// ---------------------------------------------------------------------------
// Host launcher
// ---------------------------------------------------------------------------
extern "C" void kernel_launch(void* const* d_in, const int* in_sizes, int n_in,
                              void* d_out, int out_size, void* d_ws, size_t ws_size,
                              hipStream_t stream) {
    const float* state_in = (const float*)d_in[0];
    const float* time_sig = (const float*)d_in[1];
    const float* pos_sig  = (const float*)d_in[2];
    const float* w_p      = (const float*)d_in[3];
    const float* b_p      = (const float*)d_in[4];
    const float* W1       = (const float*)d_in[5];
    const float* b1       = (const float*)d_in[6];
    const float* W2       = (const float*)d_in[7];
    const float* b2       = (const float*)d_in[8];
    float* out = (float*)d_out;

    constexpr int B = 8, S = 1024, D = 1024, DFF = 4096, HOPS = 6;
    constexpr int M = B * S;              // 8192 tokens

    char* ws = (char*)d_ws;
    size_t off = 0;
    auto walloc = [&](size_t bytes) -> void* {
        void* p = ws + off;
        off += (bytes + 255) & ~(size_t)255;
        return p;
    };
    float*  st   = (float*) walloc((size_t)M * D * 4);     // working state
    __bf16* sbf  = (__bf16*)walloc((size_t)M * D * 2);     // s in bf16
    __bf16* hbf  = (__bf16*)walloc((size_t)M * DFF * 2);   // hidden in bf16
    __bf16* w1t  = (__bf16*)walloc((size_t)D * DFF * 2);   // W1^T bf16 [DFF][D]
    __bf16* w2t  = (__bf16*)walloc((size_t)D * DFF * 2);   // W2^T bf16 [D][DFF]
    float*  hp   = (float*) walloc(M * 4);
    float*  rem  = (float*) walloc(M * 4);
    float*  nu   = (float*) walloc(M * 4);
    float*  uw   = (float*) walloc(M * 4);
    int*    flags= (int*)   walloc(64);

    init_big_kernel<<<(M * D) / 256, 256, 0, stream>>>(out, st, state_in, M * D);
    init_small_kernel<<<M / 256, 256, 0, stream>>>(hp, rem, nu, uw, flags, M, HOPS);
    convert_transpose_kernel<<<dim3(DFF / 32, D / 32), dim3(32, 8), 0, stream>>>(W1, w1t, D, DFF);
    convert_transpose_kernel<<<dim3(D / 32, DFF / 32), dim3(32, 8), 0, stream>>>(W2, w2t, DFF, D);

    for (int step = 0; step < HOPS; ++step) {
        int* flag = flags + step;
        gate_kernel<<<M / 256, 256, 0, stream>>>(hp, nu, flag, M);
        p_halt_kernel<<<M / 8, 256, 0, stream>>>(st, time_sig, pos_sig + (size_t)step * D,
                                                 w_p, b_p, hp, rem, nu, uw, sbf, flag);
        gemm_bf16_wmma<1><<<dim3(DFF / 128, M / 128), 256, 0, stream>>>(
            sbf, w1t, M, DFF, D, b1, hbf, nullptr, nullptr, nullptr, flag);
        gemm_bf16_wmma<2><<<dim3(D / 128, M / 128), 256, 0, stream>>>(
            hbf, w2t, M, D, DFF, b2, nullptr, out, st, uw, flag);
    }

    finalize_kernel<<<M / 256, 256, 0, stream>>>(rem, nu, out + (size_t)M * D, M);
}
